// TFXLMMultiHeadAttention_55327768707804
// MI455X (gfx1250) — compile-verified, matrix operand order checked by reading, not run on hardware
//
#include <hip/hip_runtime.h>
#include <hip/hip_bf16.h>

typedef __attribute__((ext_vector_type(16))) _Float16 v16h;
typedef __attribute__((ext_vector_type(8)))  _Float16 v8h;
typedef __attribute__((ext_vector_type(4)))  _Float16 v4h;
typedef __attribute__((ext_vector_type(8)))  float    v8f;
typedef __attribute__((ext_vector_type(4)))  float    v4f;

#define BS_   2
#define QLEN_ 2048
#define DIM_  1024
#define H_    16
#define DH_   64
#define MTOT  (BS_ * QLEN_)   // 4096

// ---------------------------------------------------------------------------
// WMMA helpers
// ---------------------------------------------------------------------------
__device__ __forceinline__ v8f wmma_f16(v16h a, v16h b, v8f c) {
  // D = A(16x32 f16) * B(32x16 f16) + C(16x16 f32)
  return __builtin_amdgcn_wmma_f32_16x16x32_f16(
      /*neg_a=*/false, a, /*neg_b=*/false, b,
      /*c_mod=*/(short)0, c, /*reuse_a=*/false, /*reuse_b=*/false);
}

// Load a 16x32 f16 fragment (A operand: rows row-major; B operand: columns
// stored row-major, i.e. BT[n][k]).  Layout per ISA 7.12.2:
//   lane l (0-15):  row=l, halves k0..k0+7 and k0+16..k0+23 with k0=0
//   lane l (16-31): row=l-16, same but k0=8
__device__ __forceinline__ v16h load_frag(const _Float16* __restrict__ p, int stride) {
  const int lane = threadIdx.x & 31;
  const int row  = lane & 15;
  const int k0   = (lane >> 4) << 3;           // 0 or 8
  const _Float16* q = p + row * stride + k0;
  v8h lo = *(const v8h*)(q);
  v8h hi = *(const v8h*)(q + 16);
  return __builtin_shufflevector(lo, hi, 0,1,2,3,4,5,6,7,8,9,10,11,12,13,14,15);
}

// Async global->LDS copy, 16 bytes per lane (GLOBAL_LOAD_ASYNC_TO_LDS_B128,
// GV addressing: VDST = per-lane LDS byte address, VADDR = 64-bit global).
// Tracked by ASYNCcnt.
__device__ __forceinline__ void async_copy_b128(const _Float16* gptr,
                                                _Float16* lptr) {
  const unsigned long long ga = (unsigned long long)(uintptr_t)gptr;
  const unsigned int       la = (unsigned int)(uintptr_t)lptr;
  asm volatile("global_load_async_to_lds_b128 %0, %1, off"
               :
               : "v"(la), "v"(ga)
               : "memory");
}

// ---------------------------------------------------------------------------
// f32 -> f16 elementwise convert (x -> xh)
// ---------------------------------------------------------------------------
__global__ void f32_to_f16_kernel(const float* __restrict__ in,
                                  _Float16* __restrict__ out, int n) {
  int i = (blockIdx.x * blockDim.x + threadIdx.x) * 4;
  if (i + 3 < n) {
    v4f v = *(const v4f*)(in + i);
    v4h h;
    h[0] = (_Float16)v[0]; h[1] = (_Float16)v[1];
    h[2] = (_Float16)v[2]; h[3] = (_Float16)v[3];
    *(v4h*)(out + i) = h;
  }
}

// ---------------------------------------------------------------------------
// W (K x N f32, row-major) -> WT (N x K f16)  via LDS tile transpose
// ---------------------------------------------------------------------------
__global__ void transpose_to_f16_kernel(const float* __restrict__ W,
                                        _Float16* __restrict__ WT) {
  __shared__ float tile[32][33];
  const int kb = blockIdx.x * 32;
  const int nb = blockIdx.y * 32;
  const int tx = threadIdx.x;       // 0..31
  const int ty = threadIdx.y;       // 0..7
  #pragma unroll
  for (int i = ty; i < 32; i += 8)
    tile[i][tx] = W[(kb + i) * DIM_ + nb + tx];
  __syncthreads();
  #pragma unroll
  for (int i = ty; i < 32; i += 8)
    WT[(nb + i) * DIM_ + kb + tx] = (_Float16)tile[tx][i];
}

// ---------------------------------------------------------------------------
// GEMM: C[M=4096][N=1024] = A[M][K=1024] * BT[N][K]^T, + bias, * scale
// MODE 0: f16 out, head layout  [b][h][q][d]      (Q and K)
// MODE 1: f16 out, transposed   [b][h][d][q]      (V, A-operand of P*V)
// MODE 2: f32 out, row-major    [m][n]            (final output)
// Block = 256 threads (8 waves); block tile 128x128; wave tile 64x32.
// ---------------------------------------------------------------------------
template <int MODE>
__global__ void __launch_bounds__(256)
gemm16_kernel(const _Float16* __restrict__ A, const _Float16* __restrict__ BT,
              const float* __restrict__ bias, void* __restrict__ outp,
              float scale) {
  const int lane = threadIdx.x & 31;
  const int wave = threadIdx.x >> 5;
  const int wm = wave & 1;
  const int wn = wave >> 1;
  const int m0 = blockIdx.x * 128 + wm * 64;
  const int n0 = blockIdx.y * 128 + wn * 32;

  v8f zero8 = {0.f, 0.f, 0.f, 0.f, 0.f, 0.f, 0.f, 0.f};
  v8f acc[4][2];
  #pragma unroll
  for (int i = 0; i < 4; ++i)
    #pragma unroll
    for (int j = 0; j < 2; ++j) acc[i][j] = zero8;

  for (int k0 = 0; k0 < DIM_; k0 += 32) {
    v16h a[4], b[2];
    #pragma unroll
    for (int i = 0; i < 4; ++i)
      a[i] = load_frag(A + (size_t)(m0 + i * 16) * DIM_ + k0, DIM_);
    #pragma unroll
    for (int j = 0; j < 2; ++j)
      b[j] = load_frag(BT + (size_t)(n0 + j * 16) * DIM_ + k0, DIM_);
    // Pull the next K-slab toward the caches (global_prefetch_b8).
    if (k0 + 128 < DIM_) {
      __builtin_prefetch(A + (size_t)m0 * DIM_ + k0 + 128, 0, 1);
      __builtin_prefetch(BT + (size_t)n0 * DIM_ + k0 + 128, 0, 1);
    }
    #pragma unroll
    for (int i = 0; i < 4; ++i)
      #pragma unroll
      for (int j = 0; j < 2; ++j)
        acc[i][j] = wmma_f16(a[i], b[j], acc[i][j]);
  }

  const int ln = lane & 15;
  const int lh = (lane >> 4) << 3;
  #pragma unroll
  for (int i = 0; i < 4; ++i) {
    #pragma unroll
    for (int j = 0; j < 2; ++j) {
      const int n = n0 + j * 16 + ln;
      const float bv = bias[n];
      #pragma unroll
      for (int r = 0; r < 8; ++r) {
        const int m = m0 + i * 16 + lh + r;
        const float val = (acc[i][j][r] + bv) * scale;
        if (MODE == 0) {
          const int bb = m >> 11, q = m & (QLEN_ - 1);
          const int hh = n >> 6, d = n & (DH_ - 1);
          ((_Float16*)outp)[(((size_t)(bb * H_ + hh) * QLEN_) + q) * DH_ + d] =
              (_Float16)val;
        } else if (MODE == 1) {
          const int bb = m >> 11, q = m & (QLEN_ - 1);
          const int hh = n >> 6, d = n & (DH_ - 1);
          ((_Float16*)outp)[(((size_t)(bb * H_ + hh) * DH_) + d) * QLEN_ + q] =
              (_Float16)val;
        } else {
          ((float*)outp)[(size_t)m * DIM_ + n] = val;
        }
      }
    }
  }
}

// ---------------------------------------------------------------------------
// Flash attention. One wave per 16-query tile (8 tiles / block), all waves of
// a block share the same (b,h).  K/V key-blocks of 32 are staged in LDS once
// per block via double-buffered async global->LDS copies (ASYNCcnt), so each
// wave's WMMA fragment reads are ds_load_b128 hits instead of 8x-redundant L2
// traffic, and the next block's copy overlaps the current block's WMMAs.
//
// Transposed scores S^T = K*Q^T keep softmax reductions lane-local + one
// xor-16 shuffle; the P^T C-layout maps lane-for-lane onto the B-fragment of
// the ctx^T = Vt * P^T WMMA.
// ---------------------------------------------------------------------------
__global__ void __launch_bounds__(256)
attn_kernel(const _Float16* __restrict__ Qh, const _Float16* __restrict__ Kh,
            const _Float16* __restrict__ Vt, const float* __restrict__ mask,
            _Float16* __restrict__ ctx) {
  __shared__ __align__(16) _Float16 sK[2][32 * 64];   // [buf][kk][d]
  __shared__ __align__(16) _Float16 sV[2][64 * 32];   // [buf][d][kk]

  const int tid  = threadIdx.x;
  const int lane = tid & 31;
  const int wave = tid >> 5;
  const int bh = blockIdx.x;            // 0..31
  const int b  = bh >> 4;
  const int h  = bh & 15;
  const int q0 = (blockIdx.y * 8 + wave) * 16;
  const int lh = (lane >> 4) << 3;

  const _Float16* Qb = Qh + (size_t)bh * QLEN_ * DH_;
  const _Float16* Kb = Kh + (size_t)bh * QLEN_ * DH_;
  const _Float16* Vb = Vt + (size_t)bh * DH_ * QLEN_;

  // Q^T B-fragments (d = K-dim), resident for the whole key stream.
  const v16h qf0 = load_frag(Qb + (size_t)q0 * DH_ + 0,  DH_);
  const v16h qf1 = load_frag(Qb + (size_t)q0 * DH_ + 32, DH_);

  v8f zero8 = {0.f, 0.f, 0.f, 0.f, 0.f, 0.f, 0.f, 0.f};
  v8f cacc[4];
  #pragma unroll
  for (int dt = 0; dt < 4; ++dt) cacc[dt] = zero8;
  float mrun = -3.0e38f;
  float lrun = 0.0f;

  // Cooperative staging: K tile is one contiguous 4KB slab (tid*8 halves);
  // V tile rows are 64B slices of Vt rows (row = tid>>2, chunk = tid&3).
  const int vrow = tid >> 2;
  const int vcol = (tid & 3) * 8;

  // Prologue: stage key block 0 into buffer 0.
  async_copy_b128(Kb + (size_t)0 * DH_ + tid * 8, &sK[0][tid * 8]);
  async_copy_b128(Vb + (size_t)vrow * QLEN_ + 0 + vcol, &sV[0][vrow * 32 + vcol]);

  const int NSTEP = QLEN_ / 32;   // 64
  for (int it = 0; it < NSTEP; ++it) {
    const int kk  = it * 32;
    const int cur = it & 1;

    if (it + 1 < NSTEP) {
      const int nkk = kk + 32;
      const int nb_ = cur ^ 1;
      async_copy_b128(Kb + (size_t)nkk * DH_ + tid * 8, &sK[nb_][tid * 8]);
      async_copy_b128(Vb + (size_t)vrow * QLEN_ + nkk + vcol,
                      &sV[nb_][vrow * 32 + vcol]);
      asm volatile("s_wait_asynccnt 0x2" ::: "memory");  // current buf landed
    } else {
      asm volatile("s_wait_asynccnt 0x0" ::: "memory");
    }
    __syncthreads();   // current buffer visible to all waves

    // S^T tiles from LDS: rows = keys kk..kk+15 (s0), kk+16..kk+31 (s1).
    v8f s0 = zero8, s1 = zero8;
    s0 = wmma_f16(load_frag(&sK[cur][0 * 64 + 0],   64), qf0, s0);
    s0 = wmma_f16(load_frag(&sK[cur][0 * 64 + 32],  64), qf1, s0);
    s1 = wmma_f16(load_frag(&sK[cur][16 * 64 + 0],  64), qf0, s1);
    s1 = wmma_f16(load_frag(&sK[cur][16 * 64 + 32], 64), qf1, s1);

    // Additive key mask: s -= 1e30 * (1 - m[key])
    const float* mp = mask + b * QLEN_ + kk + lh;
    v4f m0a = *(const v4f*)(mp);
    v4f m0b = *(const v4f*)(mp + 4);
    v4f m1a = *(const v4f*)(mp + 16);
    v4f m1b = *(const v4f*)(mp + 20);
    #pragma unroll
    for (int r = 0; r < 4; ++r) {
      s0[r]     += -1e30f * (1.0f - m0a[r]);
      s0[r + 4] += -1e30f * (1.0f - m0b[r]);
      s1[r]     += -1e30f * (1.0f - m1a[r]);
      s1[r + 4] += -1e30f * (1.0f - m1b[r]);
    }

    // Column (per-q) max over the 32 keys of this step.
    float tmax = s0[0];
    #pragma unroll
    for (int r = 1; r < 8; ++r) tmax = fmaxf(tmax, s0[r]);
    #pragma unroll
    for (int r = 0; r < 8; ++r) tmax = fmaxf(tmax, s1[r]);
    tmax = fmaxf(tmax, __shfl_xor(tmax, 16, 32));

    const float mnew  = fmaxf(mrun, tmax);
    const float alpha = __expf(mrun - mnew);
    #pragma unroll
    for (int dt = 0; dt < 4; ++dt)
      #pragma unroll
      for (int r = 0; r < 8; ++r) cacc[dt][r] *= alpha;
    lrun *= alpha;
    mrun = mnew;

    // p = exp(s - mnew); pack straight into the B-fragment (lane-local!).
    v16h pf;
    float tsum = 0.0f;
    #pragma unroll
    for (int r = 0; r < 8; ++r) {
      const float p = __expf(s0[r] - mnew);
      tsum += p;
      pf[r] = (_Float16)p;
    }
    #pragma unroll
    for (int r = 0; r < 8; ++r) {
      const float p = __expf(s1[r] - mnew);
      tsum += p;
      pf[8 + r] = (_Float16)p;
    }
    tsum += __shfl_xor(tsum, 16, 32);
    lrun += tsum;

    // ctx^T[d][q] += Vt[d][kk..kk+31] * P^T   (V fragments from LDS)
    #pragma unroll
    for (int dt = 0; dt < 4; ++dt)
      cacc[dt] = wmma_f16(load_frag(&sV[cur][dt * 16 * 32], 32), pf, cacc[dt]);

    __syncthreads();   // all waves done with buf[cur] before it is re-filled
  }

  // Normalize and write ctx to [b][q][h*64 + d] (f16) for the output GEMM.
  const float inv = 1.0f / lrun;
  const int q = q0 + (lane & 15);
  _Float16* cp = ctx + ((size_t)(b * QLEN_ + q)) * DIM_ + h * DH_;
  #pragma unroll
  for (int dt = 0; dt < 4; ++dt)
    #pragma unroll
    for (int r = 0; r < 8; ++r)
      cp[dt * 16 + lh + r] = (_Float16)(cacc[dt][r] * inv);
}

// ---------------------------------------------------------------------------
// Launch
// ---------------------------------------------------------------------------
extern "C" void kernel_launch(void* const* d_in, const int* in_sizes, int n_in,
                              void* d_out, int out_size, void* d_ws,
                              size_t ws_size, hipStream_t stream) {
  const float* x    = (const float*)d_in[0];
  const float* mask = (const float*)d_in[1];
  const float* Wq   = (const float*)d_in[2];
  const float* bq   = (const float*)d_in[3];
  const float* Wk   = (const float*)d_in[4];
  const float* bk   = (const float*)d_in[5];
  const float* Wv   = (const float*)d_in[6];
  const float* bv   = (const float*)d_in[7];
  const float* Wo   = (const float*)d_in[8];
  const float* bo   = (const float*)d_in[9];

  _Float16* ws   = (_Float16*)d_ws;
  _Float16* xh   = ws;                                   // 4096*1024
  _Float16* wqT  = xh  + (size_t)MTOT * DIM_;
  _Float16* wkT  = wqT + (size_t)DIM_ * DIM_;
  _Float16* wvT  = wkT + (size_t)DIM_ * DIM_;
  _Float16* woT  = wvT + (size_t)DIM_ * DIM_;
  _Float16* Qh   = woT + (size_t)DIM_ * DIM_;            // [b][h][q][d]
  _Float16* Kh   = Qh  + (size_t)MTOT * DIM_;            // [b][h][q][d]
  _Float16* Vt   = Kh  + (size_t)MTOT * DIM_;            // [b][h][d][q]
  _Float16* ctxh = Vt  + (size_t)MTOT * DIM_;            // [b][q][dim]

  // 1) x -> f16
  f32_to_f16_kernel<<<(MTOT * DIM_) / 1024, 256, 0, stream>>>(x, xh,
                                                              MTOT * DIM_);
  // 2) weight transposes -> f16 [n][k]
  dim3 tb(32, 8), tg(DIM_ / 32, DIM_ / 32);
  transpose_to_f16_kernel<<<tg, tb, 0, stream>>>(Wq, wqT);
  transpose_to_f16_kernel<<<tg, tb, 0, stream>>>(Wk, wkT);
  transpose_to_f16_kernel<<<tg, tb, 0, stream>>>(Wv, wvT);
  transpose_to_f16_kernel<<<tg, tb, 0, stream>>>(Wo, woT);

  // 3) projections (Q scaled by 1/sqrt(DH)=0.125 after bias, per reference)
  dim3 gg(MTOT / 128, DIM_ / 128);
  gemm16_kernel<0><<<gg, 256, 0, stream>>>(xh, wqT, bq, (void*)Qh, 0.125f);
  gemm16_kernel<0><<<gg, 256, 0, stream>>>(xh, wkT, bk, (void*)Kh, 1.0f);
  gemm16_kernel<1><<<gg, 256, 0, stream>>>(xh, wvT, bv, (void*)Vt, 1.0f);

  // 4) attention
  dim3 ag(BS_ * H_, QLEN_ / (8 * 16));
  attn_kernel<<<ag, 256, 0, stream>>>(Qh, Kh, Vt, mask, ctxh);

  // 5) output projection -> f32 d_out
  gemm16_kernel<2><<<gg, 256, 0, stream>>>(ctxh, woT, bo, d_out, 1.0f);
}
